// Cgc_block_44418551775904
// MI455X (gfx1250) — compile-verified, matrix operand order checked by reading, not run on hardware
//
#include <hip/hip_runtime.h>
#include <hip/hip_bf16.h>

typedef __attribute__((ext_vector_type(16))) __bf16          v16bf;
typedef __attribute__((ext_vector_type(8)))  float           v8f;

#define NODE_DIM 32
#define EDGE_DIM 16
#define ZDIM     80   // 2*NODE_DIM + EDGE_DIM
#define TILES_PER_WAVE 8

// ---------------------------------------------------------------------------
// Pre-swizzle lin_f_w / lin_s_w (80x32 f32) into bf16 WMMA B-fragments.
// Fragment index: frag = ((m*3 + c)*2 + t), layout [frag][lane][16 bf16].
// Lane half 0 holds K = {0..7, 16..23}+32c, half 1 holds K = {8..15, 24..31}+32c
// (mirrors the documented 16-bit operand VGPR striping). N = t*16 + lane%16.
// ---------------------------------------------------------------------------
__global__ void frag_setup_kernel(const float* __restrict__ wf,
                                  const float* __restrict__ wsrc,
                                  __bf16* __restrict__ frags) {
    int id = blockIdx.x * blockDim.x + threadIdx.x;
    if (id >= 12 * 32) return;
    int lane = id & 31;
    int fi   = id >> 5;           // 0..11
    int t = fi & 1;
    int c = (fi >> 1) % 3;
    int m = fi / 6;
    const float* W = m ? wsrc : wf;
    int half = lane >> 4;
    int N    = t * 16 + (lane & 15);
    __bf16* dst = frags + (size_t)id * 16;
#pragma unroll
    for (int j = 0; j < 16; ++j) {
        int kk = c * 32 + ((j < 8) ? (j + half * 8) : (16 + (j - 8) + half * 8));
        float v = (kk < ZDIM) ? W[kk * NODE_DIM + N] : 0.0f;
        dst[j] = (__bf16)v;       // hardware f32->bf16 cvt
    }
}

__global__ void zero_kernel(float* __restrict__ p, long long n) {
    long long i = (long long)blockIdx.x * blockDim.x + threadIdx.x;
    if (i < n) p[i] = 0.0f;
}

// ---------------------------------------------------------------------------
// Edge kernel: each wave owns TILES_PER_WAVE consecutive 16-edge tiles.
// All 12 bf16 B-fragments (2 matrices x 3 K-chunks x 2 N-tiles) live in
// registers for the whole wave (launch_bounds relaxes the VGPR budget so
// nothing spills); the inner tile loop only waits on the per-edge A gathers.
// 12 v_wmma_f32_16x16x32_bf16 per 16-edge tile.
// ---------------------------------------------------------------------------
__global__ void __launch_bounds__(256, 1)
edge_kernel(const float* __restrict__ x,
            const int*   __restrict__ ei,     // [2][E] src,dst
            const float* __restrict__ ea,     // [E][16]
            const float* __restrict__ ew,     // [E]
            const __bf16* __restrict__ frags,
            const float* __restrict__ lin_f_b,
            const float* __restrict__ lin_s_b,
            float* __restrict__ agg,
            int E, int numTiles) {
    const int lane  = threadIdx.x & 31;
    const int waveG = blockIdx.x * (blockDim.x >> 5) + (threadIdx.x >> 5);
    const int tile0 = waveG * TILES_PER_WAVE;
    if (tile0 >= numTiles) return;                // wave-uniform: EXEC stays full

    const int eloc = lane & 15;
    const int half = lane >> 4;
    const int h8   = half * 8;
    const int nl   = lane & 15;

    // Hoist all tile-invariant operands into registers.
    const v16bf* fb = reinterpret_cast<const v16bf*>(frags);
    const v16bf bF00 = fb[((0*3 + 0)*2 + 0)*32 + lane];
    const v16bf bF01 = fb[((0*3 + 0)*2 + 1)*32 + lane];
    const v16bf bF10 = fb[((0*3 + 1)*2 + 0)*32 + lane];
    const v16bf bF11 = fb[((0*3 + 1)*2 + 1)*32 + lane];
    const v16bf bF20 = fb[((0*3 + 2)*2 + 0)*32 + lane];
    const v16bf bF21 = fb[((0*3 + 2)*2 + 1)*32 + lane];
    const v16bf bS00 = fb[((1*3 + 0)*2 + 0)*32 + lane];
    const v16bf bS01 = fb[((1*3 + 0)*2 + 1)*32 + lane];
    const v16bf bS10 = fb[((1*3 + 1)*2 + 0)*32 + lane];
    const v16bf bS11 = fb[((1*3 + 1)*2 + 1)*32 + lane];
    const v16bf bS20 = fb[((1*3 + 2)*2 + 0)*32 + lane];
    const v16bf bS21 = fb[((1*3 + 2)*2 + 1)*32 + lane];
    const float bf0 = lin_f_b[nl], bf1 = lin_f_b[16 + nl];
    const float bs0 = lin_s_b[nl], bs1 = lin_s_b[16 + nl];

    for (int it = 0; it < TILES_PER_WAVE; ++it) {
        const int tile = tile0 + it;
        if (tile >= numTiles) break;              // wave-uniform

        int e = tile * 16 + eloc;
        int eClamp = (e < E) ? e : (E - 1);       // tail edges neutralized by w=0
        const int srcI = ei[eClamp];
        const int dstI = ei[E + eClamp];

        v8f accF0 = {}, accF1 = {}, accS0 = {}, accS1 = {};

        // ---- chunk 0: z[:,0:32) = x[dst] ----
        {
            const float* row = x + (size_t)dstI * NODE_DIM;
            float4 p0 = *reinterpret_cast<const float4*>(row + h8);
            float4 p1 = *reinterpret_cast<const float4*>(row + h8 + 4);
            float4 p2 = *reinterpret_cast<const float4*>(row + 16 + h8);
            float4 p3 = *reinterpret_cast<const float4*>(row + 16 + h8 + 4);
            v16bf a;
            a[0]=(__bf16)p0.x; a[1]=(__bf16)p0.y; a[2]=(__bf16)p0.z; a[3]=(__bf16)p0.w;
            a[4]=(__bf16)p1.x; a[5]=(__bf16)p1.y; a[6]=(__bf16)p1.z; a[7]=(__bf16)p1.w;
            a[8]=(__bf16)p2.x; a[9]=(__bf16)p2.y; a[10]=(__bf16)p2.z; a[11]=(__bf16)p2.w;
            a[12]=(__bf16)p3.x; a[13]=(__bf16)p3.y; a[14]=(__bf16)p3.z; a[15]=(__bf16)p3.w;
            accF0 = __builtin_amdgcn_wmma_f32_16x16x32_bf16(false, a, false, bF00, (short)0, accF0, false, false);
            accF1 = __builtin_amdgcn_wmma_f32_16x16x32_bf16(false, a, false, bF01, (short)0, accF1, false, false);
            accS0 = __builtin_amdgcn_wmma_f32_16x16x32_bf16(false, a, false, bS00, (short)0, accS0, false, false);
            accS1 = __builtin_amdgcn_wmma_f32_16x16x32_bf16(false, a, false, bS01, (short)0, accS1, false, false);
        }
        // ---- chunk 1: z[:,32:64) = x[src] ----
        {
            const float* row = x + (size_t)srcI * NODE_DIM;
            float4 p0 = *reinterpret_cast<const float4*>(row + h8);
            float4 p1 = *reinterpret_cast<const float4*>(row + h8 + 4);
            float4 p2 = *reinterpret_cast<const float4*>(row + 16 + h8);
            float4 p3 = *reinterpret_cast<const float4*>(row + 16 + h8 + 4);
            v16bf a;
            a[0]=(__bf16)p0.x; a[1]=(__bf16)p0.y; a[2]=(__bf16)p0.z; a[3]=(__bf16)p0.w;
            a[4]=(__bf16)p1.x; a[5]=(__bf16)p1.y; a[6]=(__bf16)p1.z; a[7]=(__bf16)p1.w;
            a[8]=(__bf16)p2.x; a[9]=(__bf16)p2.y; a[10]=(__bf16)p2.z; a[11]=(__bf16)p2.w;
            a[12]=(__bf16)p3.x; a[13]=(__bf16)p3.y; a[14]=(__bf16)p3.z; a[15]=(__bf16)p3.w;
            accF0 = __builtin_amdgcn_wmma_f32_16x16x32_bf16(false, a, false, bF10, (short)0, accF0, false, false);
            accF1 = __builtin_amdgcn_wmma_f32_16x16x32_bf16(false, a, false, bF11, (short)0, accF1, false, false);
            accS0 = __builtin_amdgcn_wmma_f32_16x16x32_bf16(false, a, false, bS10, (short)0, accS0, false, false);
            accS1 = __builtin_amdgcn_wmma_f32_16x16x32_bf16(false, a, false, bS11, (short)0, accS1, false, false);
        }
        // ---- chunk 2: z[:,64:80) = edge_attr, K 80..95 zero-padded ----
        {
            const float* row = ea + (size_t)eClamp * EDGE_DIM;
            float4 p0 = *reinterpret_cast<const float4*>(row + h8);
            float4 p1 = *reinterpret_cast<const float4*>(row + h8 + 4);
            v16bf a;
            a[0]=(__bf16)p0.x; a[1]=(__bf16)p0.y; a[2]=(__bf16)p0.z; a[3]=(__bf16)p0.w;
            a[4]=(__bf16)p1.x; a[5]=(__bf16)p1.y; a[6]=(__bf16)p1.z; a[7]=(__bf16)p1.w;
            a[8]=(__bf16)0.0f; a[9]=(__bf16)0.0f; a[10]=(__bf16)0.0f; a[11]=(__bf16)0.0f;
            a[12]=(__bf16)0.0f; a[13]=(__bf16)0.0f; a[14]=(__bf16)0.0f; a[15]=(__bf16)0.0f;
            accF0 = __builtin_amdgcn_wmma_f32_16x16x32_bf16(false, a, false, bF20, (short)0, accF0, false, false);
            accF1 = __builtin_amdgcn_wmma_f32_16x16x32_bf16(false, a, false, bF21, (short)0, accF1, false, false);
            accS0 = __builtin_amdgcn_wmma_f32_16x16x32_bf16(false, a, false, bS20, (short)0, accS0, false, false);
            accS1 = __builtin_amdgcn_wmma_f32_16x16x32_bf16(false, a, false, bS21, (short)0, accS1, false, false);
        }

        // Epilogue. D layout: lane holds column N = lane%16 (+t*16), rows M = half*8 + r.
#pragma unroll
        for (int r = 0; r < 8; ++r) {
            int M  = half * 8 + r;
            int eM = tile * 16 + M;
            float w = (eM < E) ? ew[eM] : 0.0f;
            int  d  = ei[E + ((eM < E) ? eM : (E - 1))];
            float pF0 = accF0[r] + bf0, pF1 = accF1[r] + bf1;
            float pS0 = accS0[r] + bs0, pS1 = accS1[r] + bs1;
            float g0 = 1.0f / (1.0f + __expf(-pF0));
            float g1 = 1.0f / (1.0f + __expf(-pF1));
            float v0 = (pS0 > 15.0f) ? pS0 : log1pf(__expf(pS0));
            float v1 = (pS1 > 15.0f) ? pS1 : log1pf(__expf(pS1));
            atomicAdd(&agg[(size_t)d * NODE_DIM + nl],      w * g0 * v0);
            atomicAdd(&agg[(size_t)d * NODE_DIM + 16 + nl], w * g1 * v1);
        }
    }
}

// ---------------------------------------------------------------------------
// Node kernel: wave per node; lane = feature. LayerNorm (f32) + 32x32 linear
// via shuffle-broadcast dot + residual + ELU.
// ---------------------------------------------------------------------------
__global__ void node_kernel(const float* __restrict__ x,
                            const float* __restrict__ agg,
                            const float* __restrict__ ln_g,
                            const float* __restrict__ ln_b,
                            const float* __restrict__ lin_w,
                            const float* __restrict__ lin_b,
                            float* __restrict__ out, int N) {
    const int wave = threadIdx.x >> 5;
    const int lane = threadIdx.x & 31;
    const int n = blockIdx.x * (blockDim.x >> 5) + wave;
    if (n >= N) return;
    float xv = x[(size_t)n * NODE_DIM + lane];
    float h  = agg[(size_t)n * NODE_DIM + lane] + xv;
    float s = h;
#pragma unroll
    for (int m = 16; m >= 1; m >>= 1) s += __shfl_xor(s, m);
    float mu = s * (1.0f / 32.0f);
    float d  = h - mu;
    float s2 = d * d;
#pragma unroll
    for (int m = 16; m >= 1; m >>= 1) s2 += __shfl_xor(s2, m);
    float inv = rsqrtf(s2 * (1.0f / 32.0f) + 1e-5f);
    float hn  = d * inv * ln_g[lane] + ln_b[lane];
    float acc = lin_b[lane] + xv;
#pragma unroll
    for (int k = 0; k < 32; ++k)
        acc = fmaf(__shfl(hn, k), lin_w[k * NODE_DIM + lane], acc);
    out[(size_t)n * NODE_DIM + lane] = (acc > 0.0f) ? acc : (__expf(acc) - 1.0f);
}

// ---------------------------------------------------------------------------
extern "C" void kernel_launch(void* const* d_in, const int* in_sizes, int n_in,
                              void* d_out, int out_size, void* d_ws, size_t ws_size,
                              hipStream_t stream) {
    const float* x        = (const float*)d_in[0];
    const int*   ei       = (const int*)  d_in[1];
    const float* ea       = (const float*)d_in[2];
    const float* ew       = (const float*)d_in[3];
    const float* lin_f_w  = (const float*)d_in[4];
    const float* lin_f_b  = (const float*)d_in[5];
    const float* lin_s_w  = (const float*)d_in[6];
    const float* lin_s_b  = (const float*)d_in[7];
    const float* ln_g     = (const float*)d_in[8];
    const float* ln_b     = (const float*)d_in[9];
    const float* lin_w    = (const float*)d_in[10];
    const float* lin_b    = (const float*)d_in[11];

    const int N = in_sizes[0] / NODE_DIM;
    const int E = in_sizes[3];

    // Workspace: agg [N*32 f32], then bf16 B-fragments (12 frags * 32 lanes * 16).
    float* agg = (float*)d_ws;
    size_t aggBytes = (size_t)N * NODE_DIM * sizeof(float);
    aggBytes = (aggBytes + 255) & ~(size_t)255;
    __bf16* frags = (__bf16*)((char*)d_ws + aggBytes);

    // Pass-through tuple outputs (independent of compute; queue first).
    {
        char* p = (char*)d_out + (size_t)N * NODE_DIM * sizeof(float);
        hipMemcpyAsync(p, d_in[1], (size_t)2 * E * sizeof(int),   hipMemcpyDeviceToDevice, stream);
        p += (size_t)2 * E * sizeof(int);
        hipMemcpyAsync(p, d_in[2], (size_t)E * EDGE_DIM * sizeof(float), hipMemcpyDeviceToDevice, stream);
        p += (size_t)E * EDGE_DIM * sizeof(float);
        hipMemcpyAsync(p, d_in[3], (size_t)E * sizeof(float),     hipMemcpyDeviceToDevice, stream);
    }

    long long aggN = (long long)N * NODE_DIM;
    zero_kernel<<<(int)((aggN + 255) / 256), 256, 0, stream>>>(agg, aggN);
    frag_setup_kernel<<<1, 384, 0, stream>>>(lin_f_w, lin_s_w, frags);

    int numTiles = (E + 15) / 16;
    int wavesNeeded = (numTiles + TILES_PER_WAVE - 1) / TILES_PER_WAVE;
    int edgeBlocks = (wavesNeeded + 7) / 8;   // 8 waves per 256-thread block
    edge_kernel<<<edgeBlocks, 256, 0, stream>>>(x, ei, ea, ew, frags,
                                                lin_f_b, lin_s_b, agg, E, numTiles);

    int nodeBlocks = (N + 7) / 8;
    node_kernel<<<nodeBlocks, 256, 0, stream>>>(x, agg, ln_g, ln_b, lin_w, lin_b,
                                                (float*)d_out, N);
}
